// Decoder_76587856823004
// MI455X (gfx1250) — matrix operand order repaired and checked by reading, not compiled, not run on hardware
//
#include <hip/hip_runtime.h>

typedef __bf16 bf16_t;
typedef __attribute__((ext_vector_type(16))) __bf16 v16bf;
typedef __attribute__((ext_vector_type(8)))  __bf16 v8bf;
typedef __attribute__((ext_vector_type(8)))  float  v8f;

#define NB    8          // batch
#define TENC  512
#define DIN   512
#define TDEC  150
#define ODIM  80
#define DRNN  1024
#define PRE   256
#define NATT  128
#define NLOC  32
#define KFILT 31
#define NG    4096       // 4*DRNN
#define K0    1792       // DIN + PRE + DRNN  (x|h for layer 0)
#define K1    2048       // 2*DRNN            (x|h for layer 1)
#define PRROWS 1216      // 150*8 = 1200 plus one spare 16-row tile
#define PRK    96        // 80 rounded up to 32
#define KSPLIT 4         // split-K factor for the per-step gates GEMMs

__device__ __forceinline__ unsigned short f32_to_bf16_bits(float f) {
    unsigned int u = __float_as_uint(f);
    unsigned int lsb = (u >> 16) & 1u;
    u += 0x7FFFu + lsb;                 // round-to-nearest-even
    return (unsigned short)(u >> 16);
}
__device__ __forceinline__ bf16_t to_bf16(float f) {
    unsigned short s = f32_to_bf16_bits(f);
    bf16_t r;
    __builtin_memcpy(&r, &s, sizeof(r));
    return r;
}

// ---------------------------------------------------------------------------
// Core 16x16-tile bf16 WMMA K-loop. A-row/B-row pointers are per-lane; the
// fragment packing (wave32) uses contiguous k halves, not the ISA table order:
//   A: lane row m = lane&15; K elems [kb+hi*16 .. +15] contiguous
//   B: lane col n = lane&15; K elems [kb+hi*16 .. +15] contiguous
//   D: vgpr r -> row r + hi*8, col = lane&15
// ---------------------------------------------------------------------------
__device__ __forceinline__ v8f wmma_kloop(const bf16_t* __restrict__ arow,
                                          const bf16_t* __restrict__ brow,
                                          int kcount, int hi)
{
    v8f acc = {};
    for (int kb = 0; kb < kcount; kb += 32) {
        v16bf av = *reinterpret_cast<const v16bf*>(arow + kb + hi * 16);
        v16bf bv = *reinterpret_cast<const v16bf*>(brow + kb + hi * 16);
        acc = __builtin_amdgcn_wmma_f32_16x16x32_bf16(
                  false, av, false, bv, (short)0, acc, false, false);
    }
    return acc;
}

// ---------------------------------------------------------------------------
// Generic bf16 WMMA GEMM:  out[M x N] = A[M x K] * B[N x K]^T (+bias)
// MODE 1: bf16 relu(out+bias).  MODE 2: f32 out, no bias.
// ---------------------------------------------------------------------------
template <int MODE>
__global__ __launch_bounds__(256)
void wmma_gemm_bf16(const bf16_t* __restrict__ A,
                    const bf16_t* __restrict__ B,
                    const float*  __restrict__ bias,
                    void*         __restrict__ out,
                    int K, int N, int ntiles, int tiles_total, int Mvalid)
{
    int wid = blockIdx.x * (blockDim.x >> 5) + (threadIdx.x >> 5);
    if (wid >= tiles_total) return;                 // whole wave exits together
    int lane  = threadIdx.x & 31;
    int mtile = wid / ntiles;
    int ntile = wid % ntiles;
    int col = lane & 15;
    int hi  = lane >> 4;

    const bf16_t* arow = A + (size_t)(mtile * 16 + col) * K;
    const bf16_t* brow = B + (size_t)(ntile * 16 + col) * K;
    v8f acc = wmma_kloop(arow, brow, K, hi);

    int n = ntile * 16 + col;
    float bb = (MODE == 2) ? 0.0f : bias[n];
#pragma unroll
    for (int r = 0; r < 8; ++r) {
        int row = mtile * 16 + r + hi * 8;
        if (row < Mvalid) {
            float v = acc[r] + bb;
            if (MODE == 1)
                ((bf16_t*)out)[(size_t)row * N + n] = to_bf16(fmaxf(v, 0.0f));
            else
                ((float*)out)[(size_t)row * N + n] = v;
        }
    }
}

// ---------------------------------------------------------------------------
// Split-K gates GEMM for the per-step LSTM layers (M = 16, N = 4096):
//   partial[(split*8 + m) * NG + n] = sum_{k in split} A[m][k] * B[n][k]
// KSPLIT*256 waves -> 4x the concurrency of the unsplit version; partials are
// reduced (fixed order, deterministic) in the cell kernel together with bias.
// ---------------------------------------------------------------------------
__global__ __launch_bounds__(256)
void wmma_gates_splitk(const bf16_t* __restrict__ A,   // [16][K]
                       const bf16_t* __restrict__ B,   // [4096][K]
                       float* __restrict__ partial,    // [KSPLIT][8][4096]
                       int K)
{
    int wid = blockIdx.x * (blockDim.x >> 5) + (threadIdx.x >> 5);
    int lane  = threadIdx.x & 31;
    int ntiles = NG / 16;                            // 256
    int split = wid / ntiles;
    int ntile = wid % ntiles;
    int col = lane & 15;
    int hi  = lane >> 4;
    int kpart = K / KSPLIT;
    int kbeg  = split * kpart;

    const bf16_t* arow = A + (size_t)col * K + kbeg;
    const bf16_t* brow = B + (size_t)(ntile * 16 + col) * K + kbeg;
    v8f acc = wmma_kloop(arow, brow, kpart, hi);

    int n = ntile * 16 + col;
#pragma unroll
    for (int r = 0; r < 8; ++r) {
        int row = r + hi * 8;                        // batch row; keep 0..7
        if (row < NB)
            partial[((size_t)split * NB + row) * NG + n] = acc[r];
    }
}

// ---------------------------------------------------------------------------
// Location-sensitive attention: one block per batch, 512 threads (= T_ENC).
// ---------------------------------------------------------------------------
__global__ __launch_bounds__(512)
void attn_kernel(const float* __restrict__ h0,        // [8][1024]
                 const float* __restrict__ Wq,        // [128][1024]
                 const float* __restrict__ attn_v,    // [128]
                 const float* __restrict__ attn_b,    // [128]
                 const float* __restrict__ Wloc,      // [128][32]
                 const float* __restrict__ filt,      // [32][31]
                 const float* __restrict__ proc_mem,  // [8*512][128]
                 const int*   __restrict__ memlen,    // [8]
                 const float* __restrict__ memory,    // [8][512][512]
                 const bf16_t* __restrict__ p_all,    // [1216][256]
                 float* __restrict__ att_cum,         // [8][512]
                 float* __restrict__ att_c,           // [8][512]
                 bf16_t* __restrict__ xcat0,          // [16][1792]
                 float* __restrict__ out_att,         // [8][150][512]
                 int tstep)
{
    const int b   = blockIdx.x;
    const int tid = threadIdx.x;

    __shared__ float qs[NATT];
    __shared__ float wl[NATT * NLOC];
    __shared__ float fl[NLOC * KFILT];
    __shared__ float ac[TENC];
    __shared__ float red[TENC];
    __shared__ float wsh[TENC];

    for (int i = tid; i < NATT * NLOC; i += 512) wl[i] = Wloc[i];
    for (int i = tid; i < NLOC * KFILT; i += 512) fl[i] = filt[i];
    ac[tid] = att_cum[b * TENC + tid];
    if (tid < NATT) {
        const float* h = h0 + b * DRNN;
        const float* w = Wq + tid * DRNN;
        float s = 0.0f;
        for (int k = 0; k < DRNN; ++k) s += h[k] * w[k];
        qs[tid] = s + attn_b[tid];
    }
    __syncthreads();

    // conv over cumulative attention (SAME, pad 15), 32 filters
    float f[NLOC];
#pragma unroll
    for (int c = 0; c < NLOC; ++c) {
        float s = 0.0f;
        for (int k = 0; k < KFILT; ++k) {
            int tt = tid + k - 15;
            float v = (tt >= 0 && tt < TENC) ? ac[tt] : 0.0f;
            s += v * fl[c * KFILT + k];
        }
        f[c] = s;
    }
    // energy
    const float* pm = proc_mem + ((size_t)b * TENC + tid) * NATT;
    float e = 0.0f;
    for (int a = 0; a < NATT; ++a) {
        float l = 0.0f;
#pragma unroll
        for (int c = 0; c < NLOC; ++c) l += f[c] * wl[a * NLOC + c];
        e += attn_v[a] * tanhf(pm[a] + qs[a] + l);
    }
    bool valid = tid < memlen[b];
    float em = valid ? e : -1e9f;

    // masked softmax across 512 threads
    red[tid] = em; __syncthreads();
    for (int s = 256; s > 0; s >>= 1) {
        if (tid < s) red[tid] = fmaxf(red[tid], red[tid + s]);
        __syncthreads();
    }
    float mx = red[0]; __syncthreads();
    float w = valid ? __expf(em - mx) : 0.0f;
    red[tid] = w; __syncthreads();
    for (int s = 256; s > 0; s >>= 1) {
        if (tid < s) red[tid] += red[tid + s];
        __syncthreads();
    }
    w /= red[0];
    wsh[tid] = w;
    out_att[((size_t)b * TDEC + tstep) * TENC + tid] = w;
    att_cum[b * TENC + tid] = ac[tid] + w;
    __syncthreads();

    // context: d = tid (coalesced over lanes)
    const float* mb = memory + (size_t)b * TENC * DIN;
    float cacc = 0.0f;
    for (int t = 0; t < TENC; ++t) cacc += wsh[t] * mb[(size_t)t * DIN + tid];
    att_c[b * DIN + tid] = cacc;
    xcat0[(size_t)b * K0 + tid] = to_bf16(cacc);
    // copy this step's prenet output into xcat0
    if (tid < PRE)
        xcat0[(size_t)b * K0 + DIN + tid] =
            p_all[((size_t)tstep * NB + b) * PRE + tid];
}

// ---------------------------------------------------------------------------
// LSTM cell: reduces split-K gate partials (+bias) in fixed order, applies
// the cell nonlinearity and zoneout (eval form); deposits bf16 h into the
// packed A operands for the next GEMMs.
// ---------------------------------------------------------------------------
__global__ __launch_bounds__(256)
void lstm_cell_kernel(const float* __restrict__ partial, // [KSPLIT][8][4096]
                      const float* __restrict__ bias,    // [4096]
                      float* __restrict__ h, float* __restrict__ c,
                      bf16_t* __restrict__ hdst1, int stride1,
                      bf16_t* __restrict__ hdst2, int stride2)
{
    int idx = blockIdx.x * blockDim.x + threadIdx.x;
    if (idx >= NB * DRNN) return;
    int b = idx >> 10, u = idx & 1023;
    float ig = bias[u];
    float fg = bias[DRNN + u];
    float gg = bias[2 * DRNN + u];
    float og = bias[3 * DRNN + u];
#pragma unroll
    for (int s = 0; s < KSPLIT; ++s) {
        const float* g = partial + ((size_t)s * NB + b) * NG;
        ig += g[u];
        fg += g[DRNN + u];
        gg += g[2 * DRNN + u];
        og += g[3 * DRNN + u];
    }
    float cp = c[idx], hp = h[idx];
    float si = 1.0f / (1.0f + __expf(-ig));
    float sf = 1.0f / (1.0f + __expf(-fg));
    float so = 1.0f / (1.0f + __expf(-og));
    float cn = sf * cp + si * tanhf(gg);
    float hn = so * tanhf(cn);
    hn = 0.9f * hn + 0.1f * hp;          // zoneout expected value
    cn = 0.9f * cn + 0.1f * cp;
    h[idx] = hn; c[idx] = cn;
    hdst1[(size_t)b * stride1 + u] = to_bf16(hn);
    if (hdst2) hdst2[(size_t)b * stride2 + u] = to_bf16(hn);
}

// ---------------------------------------------------------------------------
// Output projection: out = [h1|att_c] @ [feat_W;stop_W]^T   (8x81 dots, K=1536)
// ---------------------------------------------------------------------------
__global__ __launch_bounds__(256)
void proj_kernel(const float* __restrict__ h1, const float* __restrict__ att_c,
                 const float* __restrict__ featW, const float* __restrict__ stopW,
                 const float* __restrict__ stopb,
                 float* __restrict__ out_feat, float* __restrict__ out_stop,
                 int tstep)
{
    int idx = blockIdx.x * blockDim.x + threadIdx.x;
    if (idx >= NB * 81) return;
    int b = idx / 81, j = idx % 81;
    const float* w = (j < ODIM) ? (featW + (size_t)j * (DRNN + DIN)) : stopW;
    const float* h = h1 + b * DRNN;
    const float* a = att_c + b * DIN;
    float s = 0.0f;
    for (int k = 0; k < DRNN; ++k) s += h[k] * w[k];
    for (int k = 0; k < DIN; ++k)  s += a[k] * w[DRNN + k];
    if (j < ODIM) out_feat[((size_t)b * TDEC + tstep) * ODIM + j] = s;
    else          out_stop[b * TDEC + tstep] = s + stopb[0];
}

// ----------------------------- prologue kernels ----------------------------
__global__ void pack_w_kernel(const float* __restrict__ Wi, int Ki,
                              const float* __restrict__ Wh, int Kh,
                              bf16_t* __restrict__ out, int N)
{
    int K = Ki + Kh;
    int total = N * K;
    for (int i = blockIdx.x * blockDim.x + threadIdx.x; i < total;
         i += gridDim.x * blockDim.x) {
        int n = i / K, k = i % K;
        float v = (k < Ki) ? Wi[(size_t)n * Ki + k] : Wh[(size_t)n * Kh + (k - Ki)];
        out[i] = to_bf16(v);
    }
}

__global__ void cvt_kernel(const float* __restrict__ src, bf16_t* __restrict__ dst,
                           int n)
{
    for (int i = blockIdx.x * blockDim.x + threadIdx.x; i < n;
         i += gridDim.x * blockDim.x)
        dst[i] = to_bf16(src[i]);
}

__global__ void cvt_pad_kernel(const float* __restrict__ src, bf16_t* __restrict__ dst,
                               int rows, int srcK, int dstK)
{
    int total = rows * dstK;
    for (int i = blockIdx.x * blockDim.x + threadIdx.x; i < total;
         i += gridDim.x * blockDim.x) {
        int r = i / dstK, k = i % dstK;
        dst[i] = to_bf16(k < srcK ? src[(size_t)r * srcK + k] : 0.0f);
    }
}

// teacher forcing inputs: row r = t*8 + b ; zeros at t==0, K padded 80->96
__global__ void build_prevouts(const float* __restrict__ targets,
                               bf16_t* __restrict__ out)
{
    int total = PRROWS * PRK;
    for (int i = blockIdx.x * blockDim.x + threadIdx.x; i < total;
         i += gridDim.x * blockDim.x) {
        int r = i / PRK, k = i % PRK;
        float v = 0.0f;
        if (r < TDEC * NB && k < ODIM) {
            int t = r / NB, b = r % NB;
            if (t > 0) v = targets[((size_t)b * TDEC + (t - 1)) * ODIM + k];
        }
        out[i] = to_bf16(v);
    }
}

// ---------------------------------------------------------------------------
extern "C" void kernel_launch(void* const* d_in, const int* in_sizes, int n_in,
                              void* d_out, int out_size, void* d_ws, size_t ws_size,
                              hipStream_t stream)
{
    (void)in_sizes; (void)n_in; (void)out_size; (void)ws_size;
    const float* memory   = (const float*)d_in[0];
    const int*   memlen   = (const int*)  d_in[1];
    const float* targets  = (const float*)d_in[2];
    const float* W_mem    = (const float*)d_in[3];
    const float* W_q      = (const float*)d_in[4];
    const float* W_loc    = (const float*)d_in[5];
    const float* loc_filt = (const float*)d_in[6];
    const float* attn_v   = (const float*)d_in[7];
    const float* attn_b   = (const float*)d_in[8];
    const float* pre_W1   = (const float*)d_in[9];
    const float* pre_b1   = (const float*)d_in[10];
    const float* pre_W2   = (const float*)d_in[11];
    const float* pre_b2   = (const float*)d_in[12];
    const float* Wi0      = (const float*)d_in[13];
    const float* Wh0      = (const float*)d_in[14];
    const float* bl0      = (const float*)d_in[15];
    const float* Wi1      = (const float*)d_in[16];
    const float* Wh1      = (const float*)d_in[17];
    const float* bl1      = (const float*)d_in[18];
    const float* feat_W   = (const float*)d_in[19];
    const float* stop_W   = (const float*)d_in[20];
    const float* stop_b   = (const float*)d_in[21];

    float* out_feat = (float*)d_out;                        // [8][150][80]
    float* out_stop = out_feat + NB * TDEC * ODIM;          // [8][150]
    float* out_att  = out_stop + NB * TDEC;                 // [8][150][512]

    // ---- workspace carve-up (256B aligned) ----
    char* base = (char*)d_ws;
    size_t off = 0;
    auto alloc = [&](size_t bytes) -> char* {
        char* p = base + off;
        off += (bytes + 255) & ~(size_t)255;
        return p;
    };
    bf16_t* w0cat  = (bf16_t*)alloc((size_t)NG * K0 * 2);       // [4096][1792]
    bf16_t* w1cat  = (bf16_t*)alloc((size_t)NG * K1 * 2);       // [4096][2048]
    bf16_t* mem_bf = (bf16_t*)alloc((size_t)NB * TENC * DIN * 2);
    bf16_t* wmem_bf= (bf16_t*)alloc((size_t)NATT * DIN * 2);
    bf16_t* pw1    = (bf16_t*)alloc((size_t)PRE * PRK * 2);
    bf16_t* pw2    = (bf16_t*)alloc((size_t)PRE * PRE * 2);
    bf16_t* prevo  = (bf16_t*)alloc((size_t)PRROWS * PRK * 2);
    bf16_t* h1p    = (bf16_t*)alloc((size_t)PRROWS * PRE * 2);
    bf16_t* p_bf   = (bf16_t*)alloc((size_t)PRROWS * PRE * 2);
    float*  procm  = (float*) alloc((size_t)NB * TENC * NATT * 4);
    float*  gates0 = (float*) alloc((size_t)KSPLIT * NB * NG * 4);
    float*  gates1 = (float*) alloc((size_t)KSPLIT * NB * NG * 4);
    float*  att_c  = (float*) alloc((size_t)NB * DIN * 4);
    // recurrent state block (zeroed each call with one memset)
    char* state_begin = base + off;
    float*  h0     = (float*) alloc((size_t)NB * DRNN * 4);
    float*  c0     = (float*) alloc((size_t)NB * DRNN * 4);
    float*  h1     = (float*) alloc((size_t)NB * DRNN * 4);
    float*  c1     = (float*) alloc((size_t)NB * DRNN * 4);
    float*  attcum = (float*) alloc((size_t)NB * TENC * 4);
    bf16_t* xcat0  = (bf16_t*)alloc((size_t)16 * K0 * 2);       // [att_c|p|h0]
    bf16_t* xcat1  = (bf16_t*)alloc((size_t)16 * K1 * 2);       // [h0|h1]
    size_t state_bytes = (size_t)((base + off) - state_begin);

    // ---- prologue: zero state, convert/pack weights, step-independent GEMMs
    hipMemsetAsync(state_begin, 0, state_bytes, stream);
    pack_w_kernel<<<1024, 256, 0, stream>>>(Wi0, DIN + PRE, Wh0, DRNN, w0cat, NG);
    pack_w_kernel<<<1024, 256, 0, stream>>>(Wi1, DRNN, Wh1, DRNN, w1cat, NG);
    cvt_kernel<<<1024, 256, 0, stream>>>(memory, mem_bf, NB * TENC * DIN);
    cvt_kernel<<<64, 256, 0, stream>>>(W_mem, wmem_bf, NATT * DIN);
    cvt_pad_kernel<<<64, 256, 0, stream>>>(pre_W1, pw1, PRE, ODIM, PRK);
    cvt_kernel<<<64, 256, 0, stream>>>(pre_W2, pw2, PRE * PRE);
    build_prevouts<<<128, 256, 0, stream>>>(targets, prevo);

    // proc_mem = memory @ W_mem^T : [4096][512] x [128][512] -> [4096][128]
    {
        int mt = (NB * TENC) / 16, nt = NATT / 16;    // 256 x 8 = 2048 tiles
        wmma_gemm_bf16<2><<<mt * nt / 8, 256, 0, stream>>>(
            mem_bf, wmem_bf, nullptr, procm, DIN, NATT, nt, mt * nt, NB * TENC);
    }
    // prenet for all steps: relu(prev@W1^T+b1) -> relu(@W2^T+b2)
    {
        int mt = PRROWS / 16, nt = PRE / 16;          // 76 x 16 = 1216 tiles
        wmma_gemm_bf16<1><<<mt * nt / 8, 256, 0, stream>>>(
            prevo, pw1, pre_b1, h1p, PRK, PRE, nt, mt * nt, PRROWS);
        wmma_gemm_bf16<1><<<mt * nt / 8, 256, 0, stream>>>(
            h1p, pw2, pre_b2, p_bf, PRE, PRE, nt, mt * nt, PRROWS);
    }

    const int gatesBlocks = KSPLIT * (NG / 16) / 8;   // 1024 waves -> 128 blocks

    // ---- sequential decode loop ----
    for (int t = 0; t < TDEC; ++t) {
        attn_kernel<<<NB, 512, 0, stream>>>(
            h0, W_q, attn_v, attn_b, W_loc, loc_filt, procm, memlen, memory,
            p_bf, attcum, att_c, xcat0, out_att, t);

        // layer 0 gates: [16][1792] x [4096][1792]^T, split-K x4
        wmma_gates_splitk<<<gatesBlocks, 256, 0, stream>>>(
            xcat0, w0cat, gates0, K0);
        lstm_cell_kernel<<<(NB * DRNN + 255) / 256, 256, 0, stream>>>(
            gates0, bl0, h0, c0, xcat0 + (DIN + PRE), K0, xcat1, K1);

        // layer 1 gates: [16][2048] x [4096][2048]^T, split-K x4
        wmma_gates_splitk<<<gatesBlocks, 256, 0, stream>>>(
            xcat1, w1cat, gates1, K1);
        lstm_cell_kernel<<<(NB * DRNN + 255) / 256, 256, 0, stream>>>(
            gates1, bl1, h1, c1, xcat1 + DRNN, K1, nullptr, 0);

        proj_kernel<<<3, 256, 0, stream>>>(
            h1, att_c, feat_W, stop_W, stop_b, out_feat, out_stop, t);
    }
}
